// ModifiedLinear_60284160966842
// MI455X (gfx1250) — compile-verified, matrix operand order checked by reading, not run on hardware
//
#include <hip/hip_runtime.h>
#include <stddef.h>

// ---------------------------------------------------------------------------
// Reference: out[n] = diag( (f @ E + b) @ w ),  f:(8192,512) E:(512,4096)
//            w:(4096,8192), b:(4096,)  -> out:(8192,) f32
//
// Rewritten: out[n] = sum_k f[n,k] * T[n,k] + sum_d b[d] w[d,n]
//            T = (E @ w)^T, computed as C[8192 x 528] = A @ B with
//            A[n,d] = w[d,n], B[d,c] = E[c,d] (c<512) | b[d] (c==512) | 0.
// The diag-contraction with f happens in registers; T never hits HBM.
// Traffic: w 134 MB once, f 16.8 MB once, E L2-resident -> ~7 us floor.
//
// Precision: f32 split into bf16 hi/lo; 3 bf16 WMMAs per tile-step
// (ah*bh + ah*bl + al*bh) ~= f32 product to ~1e-5 relative, at bf16-WMMA
// rate (103 GFLOP) instead of the 8x-slower f32 16x16x4 WMMA path.
// ---------------------------------------------------------------------------

#define N_ROWS   8192
#define K_DIM    512
#define D_DIM    4096
#define CW_ROWS  528            // 512 + 1 bias col + 15 zero pad = 33 tiles
#define M_BLK    32             // rows of A per workgroup
#define A_STRIDE 40             // bf16 units per staged A row (32 + 8 pad)

typedef __attribute__((ext_vector_type(16))) __bf16 v16bf;
typedef __attribute__((ext_vector_type(8)))  __bf16 v8bf;
typedef __attribute__((ext_vector_type(4)))  __bf16 v4bf;
typedef __attribute__((ext_vector_type(8)))  float  v8f;

__device__ __forceinline__ unsigned short bf16_rne(float x) {
  unsigned u = __builtin_bit_cast(unsigned, x);
  return (unsigned short)((u + 0x7FFFu + ((u >> 16) & 1u)) >> 16);
}

__device__ __forceinline__ void split_f32(float x, unsigned short& hi, unsigned short& lo) {
  hi = bf16_rne(x);
  float hf = __builtin_bit_cast(float, (unsigned)hi << 16);
  lo = bf16_rne(x - hf);
}

// ---------------------------------------------------------------------------
// Pre-pass: split E (and bias b as row 512, zeros above) into bf16 hi/lo
// planes laid out [c][d] (d contiguous) -- exactly the B-operand order.
// Done once so the 256 GEMM workgroups don't redo the f32->bf16x2 VALU work.
// ---------------------------------------------------------------------------
__global__ __launch_bounds__(256) void split_EB_kernel(
    const float* __restrict__ E, const float* __restrict__ b,
    unsigned short* __restrict__ Bhi, unsigned short* __restrict__ Blo) {
  int idx = blockIdx.x * 256 + threadIdx.x;
  if (idx >= CW_ROWS * D_DIM) return;
  int c = idx >> 12;                 // / 4096
  int d = idx & (D_DIM - 1);
  float x = 0.0f;
  if (c < K_DIM)       x = E[idx];
  else if (c == K_DIM) x = b[d];
  unsigned short hi, lo;
  split_f32(x, hi, lo);
  Bhi[idx] = hi;
  Blo[idx] = lo;
}

// ---------------------------------------------------------------------------
// Fused GEMM (bf16x3 WMMA) + diag contraction with f.
// Block: 256 threads = 8 waves; each block owns 32 rows (m0..m0+31) and the
// full 528-column width (wave c-tiles: wave*4..wave*4+3, wave 7 adds tile 32).
// ---------------------------------------------------------------------------
__global__ __launch_bounds__(256) void fused_wmma_diag_kernel(
    const float* __restrict__ f,           // (8192, 512)
    const float* __restrict__ w,           // (4096, 8192)
    const unsigned short* __restrict__ Bh, // (528, 4096) bf16 hi bits
    const unsigned short* __restrict__ Bl, // (528, 4096) bf16 lo bits
    float* __restrict__ out) {
  __shared__ __align__(16) __bf16 Ahi[M_BLK * A_STRIDE];
  __shared__ __align__(16) __bf16 Alo[M_BLK * A_STRIDE];
  __shared__ float out_lds[M_BLK];

  const int t     = threadIdx.x;
  const int wave  = t >> 5;        // 0..7
  const int lane  = t & 31;
  const int lrow  = lane & 15;
  const int lhalf = lane >> 4;     // 0 or 1
  const int m0    = blockIdx.x * M_BLK;

  if (t < M_BLK) out_lds[t] = 0.0f;

  v8f acc[2][5];
#pragma unroll
  for (int i = 0; i < 2; ++i)
#pragma unroll
    for (int j = 0; j < 5; ++j)
      acc[i][j] = v8f{};

  const int nct = (wave == 7) ? 5 : 4;   // wave 7 also owns the bias tile

  for (int dc = 0; dc < D_DIM / 32; ++dc) {
    const int d0 = dc * 32;
    __syncthreads();   // previous iteration's fragment reads done

    // ---- stage A tile: w[(d0+d)][m0+m] -> LDS [m][d], split hi/lo --------
    // Thread owns 4 consecutive d values -> packed v4bf ds_store_b64.
    {
      const int m_l = lane;            // 0..31 (coalesced along n of w)
      const int d_b = wave * 4;        // 0,4,...,28
      v4bf hp, lp;
#pragma unroll
      for (int i = 0; i < 4; ++i) {
        int d = d_b + i;
        float x = w[(size_t)(d0 + d) * N_ROWS + m0 + m_l];
        unsigned short hi, lo;
        split_f32(x, hi, lo);
        hp[i] = __builtin_bit_cast(__bf16, hi);
        lp[i] = __builtin_bit_cast(__bf16, lo);
        // pull next chunk's A rows toward the WGP while this chunk computes
        __builtin_prefetch(&w[(size_t)(d0 + 32 + d) * N_ROWS + m0 + m_l], 0, 1);
      }
      *(v4bf*)(Ahi + m_l * A_STRIDE + d_b) = hp;
      *(v4bf*)(Alo + m_l * A_STRIDE + d_b) = lp;
    }
    __syncthreads();

    // ---- A fragments (16x32 bf16, ISA 7.12.2 interleave) -----------------
    v16bf afh[2], afl[2];
#pragma unroll
    for (int mt = 0; mt < 2; ++mt) {
      const __bf16* ra = Ahi + (mt * 16 + lrow) * A_STRIDE;
      v8bf p0 = *(const v8bf*)(ra + lhalf * 8);
      v8bf p1 = *(const v8bf*)(ra + 16 + lhalf * 8);
      afh[mt] = __builtin_shufflevector(p0, p1, 0,1,2,3,4,5,6,7,8,9,10,11,12,13,14,15);
      const __bf16* rl = Alo + (mt * 16 + lrow) * A_STRIDE;
      v8bf q0 = *(const v8bf*)(rl + lhalf * 8);
      v8bf q1 = *(const v8bf*)(rl + 16 + lhalf * 8);
      afl[mt] = __builtin_shufflevector(q0, q1, 0,1,2,3,4,5,6,7,8,9,10,11,12,13,14,15);
    }

    // ---- preload ALL B fragments (distinct registers -> loads overlap) ---
    v16bf bfh[5], bfl[5];
#pragma unroll
    for (int j = 0; j < 5; ++j) {
      if (j < nct) {
        const int ctg  = (j < 4) ? (wave * 4 + j) : 32;      // col tile index
        const size_t rbase = (size_t)(ctg * 16 + lrow) * D_DIM + d0;
        v8bf b0 = *(const v8bf*)(Bh + rbase + lhalf * 8);
        v8bf b1 = *(const v8bf*)(Bh + rbase + 16 + lhalf * 8);
        bfh[j] = __builtin_shufflevector(b0, b1, 0,1,2,3,4,5,6,7,8,9,10,11,12,13,14,15);
        v8bf c0 = *(const v8bf*)(Bl + rbase + lhalf * 8);
        v8bf c1 = *(const v8bf*)(Bl + rbase + 16 + lhalf * 8);
        bfl[j] = __builtin_shufflevector(c0, c1, 0,1,2,3,4,5,6,7,8,9,10,11,12,13,14,15);
      }
    }

    // ---- WMMA burst: 3 bf16 terms per (mt, j) tile -----------------------
#pragma unroll
    for (int j = 0; j < 5; ++j) {
      if (j < nct) {
#pragma unroll
        for (int mt = 0; mt < 2; ++mt) {
          acc[mt][j] = __builtin_amdgcn_wmma_f32_16x16x32_bf16(
              false, afh[mt], false, bfh[j], (short)0, acc[mt][j], false, false);
          acc[mt][j] = __builtin_amdgcn_wmma_f32_16x16x32_bf16(
              false, afl[mt], false, bfh[j], (short)0, acc[mt][j], false, false);
          acc[mt][j] = __builtin_amdgcn_wmma_f32_16x16x32_bf16(
              false, afh[mt], false, bfl[j], (short)0, acc[mt][j], false, false);
        }
      }
    }
  }

  // ---- epilogue: out[n] += sum_c T[n,c] * f[n,c]  (+ bias column) --------
  // C/D layout: VGPR r holds (M = r + 8*lhalf, N = lrow) of each 16x16 tile.
#pragma unroll
  for (int mt = 0; mt < 2; ++mt) {
#pragma unroll
    for (int r = 0; r < 8; ++r) {
      const int row = mt * 16 + 8 * lhalf + r;                 // 0..31
      const float* frow = f + (size_t)(m0 + row) * K_DIM;
      float partial = 0.0f;
#pragma unroll
      for (int j = 0; j < 4; ++j) {
        int col = wave * 64 + j * 16 + lrow;                   // < 512
        partial += acc[mt][j][r] * frow[col];
      }
      if (wave == 7)  // bias tile: only column 512 (lrow==0) carries weight 1
        partial += acc[mt][4][r] * ((lrow == 0) ? 1.0f : 0.0f);
      atomicAdd(&out_lds[row], partial);
    }
  }
  __syncthreads();
  if (t < M_BLK) out[m0 + t] = out_lds[t];
}

// ---------------------------------------------------------------------------
extern "C" void kernel_launch(void* const* d_in, const int* in_sizes, int n_in,
                              void* d_out, int out_size, void* d_ws, size_t ws_size,
                              hipStream_t stream) {
  (void)in_sizes; (void)n_in; (void)out_size; (void)ws_size;
  const float* f = (const float*)d_in[0];   // features (8192, 512)
  const float* w = (const float*)d_in[1];   // w        (4096, 8192)
  const float* E = (const float*)d_in[2];   // E        (512, 4096)
  const float* b = (const float*)d_in[3];   // b        (4096,)
  float* out = (float*)d_out;

  unsigned short* Bhi = (unsigned short*)d_ws;                   // 528*4096 u16
  unsigned short* Blo = Bhi + (size_t)CW_ROWS * D_DIM;           // 528*4096 u16

  const int splitN = CW_ROWS * D_DIM;
  split_EB_kernel<<<(splitN + 255) / 256, 256, 0, stream>>>(E, b, Bhi, Blo);
  fused_wmma_diag_kernel<<<N_ROWS / M_BLK, 256, 0, stream>>>(f, w, Bhi, Blo, out);
}
